// TritonWindowedSelfAttention_40724879900946
// MI455X (gfx1250) — matrix operand order repaired and checked by reading, not compiled, not run on hardware
//
#include <hip/hip_runtime.h>
#include <hip/hip_bf16.h>
#include <stdint.h>

typedef __attribute__((ext_vector_type(16))) __bf16 v16bf;
typedef __attribute__((ext_vector_type(8)))  __bf16 bf16x8;
typedef __attribute__((ext_vector_type(8)))  float  v8f;
typedef __attribute__((ext_vector_type(4)))  float  f32x4;
typedef __attribute__((ext_vector_type(4)))  uint32_t u32x4;
typedef __attribute__((ext_vector_type(8)))  uint32_t u32x8;

#define DEV static __device__ __forceinline__

// ---- problem constants (match reference) ----
constexpr int CB  = 4;       // batch
constexpr int CS  = 4096;    // seq
constexpr int CD  = 1024;    // model dim
constexpr int CH  = 16;      // heads
constexpr int CDh = 64;      // head dim
constexpr int CW  = 128;     // window
constexpr int CM  = CB * CS; // 16384 rows

DEV __bf16 f2bf(float f) {
  uint32_t u = __builtin_bit_cast(uint32_t, f);
  uint32_t r = u + 0x7FFFu + ((u >> 16) & 1u);   // round-to-nearest-even
  uint16_t s = (uint16_t)(r >> 16);
  return __builtin_bit_cast(__bf16, s);
}

DEV v16bf pack16(bf16x8 lo, bf16x8 hi) {
  union { v16bf v; bf16x8 h[2]; } u;
  u.h[0] = lo; u.h[1] = hi;
  return u.v;
}

DEV v8f wmma_bf16(v16bf a, v16bf b, v8f c) {
  return __builtin_amdgcn_wmma_f32_16x16x32_bf16(
      false, a, false, b, (short)0, c, false, false);
}

// TDM copy of a contiguous 32KB strip (16 x CD bf16) into LDS.
// D# per CDNA5 ISA 8.3/8.4: count=1, data_size=2B, tile 16384x1, type=2.
DEV void tdm_stage_strip(const __bf16* gsrc, uint32_t lds_addr) {
  uint64_t ga = (uint64_t)(uintptr_t)gsrc;
  u32x4 g0;
  g0[0] = 1u;                                   // count=1, user mode
  g0[1] = lds_addr;                             // lds_addr (bytes)
  g0[2] = (uint32_t)ga;                         // global_addr[31:0]
  g0[3] = (uint32_t)((ga >> 32) & 0x01FFFFFFu) | (2u << 30); // addr[56:32]|type=2
  u32x8 g1;
  g1[0] = (1u << 16);                           // wg_mask=0, data_size=1 (2B)
  g1[1] = 0x40000000u;                          // tensor_dim0=16384 (lo16 @ bits63:48)
  g1[2] = 0x00010000u;                          // tensor_dim0 hi=0, tensor_dim1=1
  g1[3] = 0x40000000u;                          // tile_dim0=16384
  g1[4] = 1u;                                   // tile_dim1=1, tile_dim2=0
  g1[5] = 16384u;                               // tensor_dim0_stride lo32
  g1[6] = 0u;
  g1[7] = 0u;
  asm volatile("tensor_load_to_lds %0, %1" :: "s"(g0), "s"(g1) : "memory");
  __builtin_amdgcn_s_wait_tensorcnt(0);
}

// ---------------------------------------------------------------------------
// Kernel 1: fp32 -> bf16 conversion
// ---------------------------------------------------------------------------
__global__ void cvt_f32_bf16(const float* __restrict__ src,
                             __bf16* __restrict__ dst, int n) {
  int i = (blockIdx.x * blockDim.x + threadIdx.x) * 8;
  if (i + 8 <= n) {
    f32x4 a = *(const f32x4*)(src + i);
    f32x4 b = *(const f32x4*)(src + i + 4);
    bf16x8 o;
    o[0] = f2bf(a[0]); o[1] = f2bf(a[1]); o[2] = f2bf(a[2]); o[3] = f2bf(a[3]);
    o[4] = f2bf(b[0]); o[5] = f2bf(b[1]); o[6] = f2bf(b[2]); o[7] = f2bf(b[3]);
    *(bf16x8*)(dst + i) = o;
  }
}

// ---------------------------------------------------------------------------
// One projection (MODE 0=Q scaled, 1=K, 2=V transposed).  Template with
// concrete global pointers so address-space inference keeps global_load_b128
// (a z-indexed pointer array degraded these to flat_load last round).
// ---------------------------------------------------------------------------
template <int MODE>
DEV void proj_one(const __bf16* xs, const __bf16* __restrict__ W,
                  const float* __restrict__ bias,
                  __bf16* __restrict__ OutQK, __bf16* __restrict__ OutVt,
                  int colbase, int m, int hh, int kb, int bb, int s0) {
  const __bf16* wbase = W + (size_t)(colbase + m) * CD + hh * 16;
  v8f acc[4] = {};
  for (int k0 = 0; k0 < CD; k0 += 32) {
    bf16x8 alo = *(const bf16x8*)(xs + m * CD + k0 + kb);
    bf16x8 ahi = *(const bf16x8*)(xs + m * CD + k0 + kb + 8);
    v16bf a = pack16(alo, ahi);
    __builtin_prefetch(wbase + k0 + 128, 0, 3);
#pragma unroll
    for (int t = 0; t < 4; ++t) {
      const __bf16* wrow = wbase + (size_t)t * 16 * CD + k0;
      v16bf b = pack16(*(const bf16x8*)(wrow), *(const bf16x8*)(wrow + 8));
      acc[t] = wmma_bf16(a, b, acc[t]);
    }
  }
#pragma unroll
  for (int t = 0; t < 4; ++t) {
    const int o = colbase + t * 16 + m;
    const float bv = bias[o];
    const int h = o >> 6, dh = o & 63;
    if (MODE == 2) {
      bf16x8 vv;
#pragma unroll
      for (int j = 0; j < 8; ++j) vv[j] = f2bf(acc[t][j] + bv);
      size_t base = (((size_t)bb * CH + h) * CDh + dh) * CS + s0 + hh * 8;
      *(bf16x8*)(OutVt + base) = vv;
    } else {
      const float sc = (MODE == 0) ? 0.125f : 1.0f;
#pragma unroll
      for (int j = 0; j < 8; ++j) {
        int srow = s0 + j + hh * 8;
        OutQK[(((size_t)bb * CH + h) * CS + srow) * CDh + dh] =
            f2bf((acc[t][j] + bv) * sc);
      }
    }
  }
}

// ---------------------------------------------------------------------------
// Kernel 2: fused QKV projection; X strip staged once via Tensor Data Mover.
// Block 128 thr (4 waves): wave covers 64 out-cols; grid (CM/16, CD/256)
// ---------------------------------------------------------------------------
__global__ __launch_bounds__(128) void qkv_proj(
    const __bf16* __restrict__ X,
    const __bf16* __restrict__ Wq, const __bf16* __restrict__ Wk,
    const __bf16* __restrict__ Wv,
    const float* __restrict__ bq, const float* __restrict__ bk,
    const float* __restrict__ bv,
    __bf16* __restrict__ Qs, __bf16* __restrict__ Kb, __bf16* __restrict__ Vt) {
  __shared__ __align__(16) __bf16 xs[16 * CD];   // 32 KB A-strip
  const int tid = threadIdx.x;
  const int rowtile = blockIdx.x;

  if (tid < 32)    // one wave drives the TDM DMA for the whole block
    tdm_stage_strip(X + (size_t)rowtile * 16 * CD, (uint32_t)(uintptr_t)xs);
  __syncthreads();

  const int lane = tid & 31, wave = tid >> 5;
  const int colbase = (blockIdx.y * 4 + wave) * 64;
  const int m  = lane & 15;
  const int hh = lane >> 4;
  const int kb = hh * 16;
  const int rowbase = rowtile * 16;
  const int bb = rowbase >> 12;
  const int s0 = rowbase & (CS - 1);

  proj_one<0>(xs, Wq, bq, Qs, nullptr, colbase, m, hh, kb, bb, s0);
  proj_one<1>(xs, Wk, bk, Kb, nullptr, colbase, m, hh, kb, bb, s0);
  proj_one<2>(xs, Wv, bv, nullptr, Vt, colbase, m, hh, kb, bb, s0);
}

// ---------------------------------------------------------------------------
// Kernel 3: windowed attention, one wave per 16-query tile.
// ---------------------------------------------------------------------------
__global__ __launch_bounds__(128) void attn_win(
    const __bf16* __restrict__ Qs, const __bf16* __restrict__ Kb,
    const __bf16* __restrict__ Vt, __bf16* __restrict__ Ob) {
  __shared__ __align__(16) __bf16 plds[4][16][48];
  const int tid = threadIdx.x, lane = tid & 31, wave = tid >> 5;
  const int bh = blockIdx.y, bb = bh >> 4, h = bh & 15;
  const int q0 = (blockIdx.x * 4 + wave) * 16;

  const __bf16* Qp = Qs + (size_t)bh * CS * CDh;
  const __bf16* Kp = Kb + (size_t)bh * CS * CDh;
  const __bf16* Vp = Vt + (size_t)bh * CDh * CS;

  const int m  = lane & 15;
  const int hh = lane >> 4;
  const int kb = hh * 16;

  const __bf16* qrow = Qp + (size_t)(q0 + m) * CDh;
  v16bf aq0 = pack16(*(const bf16x8*)(qrow + kb),
                     *(const bf16x8*)(qrow + kb + 8));
  v16bf aq1 = pack16(*(const bf16x8*)(qrow + 32 + kb),
                     *(const bf16x8*)(qrow + 32 + kb + 8));

  float mrow[8], tacc[8];
  v8f oacc[4] = {};
  for (int j = 0; j < 8; ++j) { mrow[j] = -1e30f; tacc[j] = 0.f; }

  __bf16 (*pw)[48] = plds[wave];

  for (int ch = 0; ch < 9; ++ch) {
    const int kstart = q0 - CW + ch * 32;

    v8f sfr[2];
#pragma unroll
    for (int sub = 0; sub < 2; ++sub) {
      int ks  = kstart + sub * 16;
      int key = ks + m;
      int keyc = key < 0 ? 0 : (key > CS - 1 ? CS - 1 : key);
      const __bf16* krow = Kp + (size_t)keyc * CDh + hh * 16;
      v16bf b0 = pack16(*(const bf16x8*)(krow),      *(const bf16x8*)(krow + 8));
      v16bf b1 = pack16(*(const bf16x8*)(krow + 32), *(const bf16x8*)(krow + 40));
      v8f sf = {};
      sf = wmma_bf16(aq0, b0, sf);
      sf = wmma_bf16(aq1, b1, sf);
#pragma unroll
      for (int j = 0; j < 8; ++j) {
        int q = q0 + j + hh * 8;
        int dpos = key - q;
        bool valid = (key >= 0) & (key < CS) & (dpos >= -CW) & (dpos <= CW);
        sf[j] = valid ? sf[j] : -1e30f;
      }
      sfr[sub] = sf;
    }

#pragma unroll
    for (int j = 0; j < 8; ++j) {
      float v = fmaxf(sfr[0][j], sfr[1][j]);
      v = fmaxf(v, __shfl_xor(v, 1));
      v = fmaxf(v, __shfl_xor(v, 2));
      v = fmaxf(v, __shfl_xor(v, 4));
      v = fmaxf(v, __shfl_xor(v, 8));
      float mn = fmaxf(mrow[j], v);
      float sc = __expf(mrow[j] - mn);
      mrow[j] = mn;
      tacc[j] *= sc;
      oacc[0][j] *= sc; oacc[1][j] *= sc; oacc[2][j] *= sc; oacc[3][j] *= sc;
      float p0 = (sfr[0][j] > -1e29f) ? __expf(sfr[0][j] - mn) : 0.f;
      float p1 = (sfr[1][j] > -1e29f) ? __expf(sfr[1][j] - mn) : 0.f;
      tacc[j] += p0 + p1;
      pw[j + hh * 8][m]      = f2bf(p0);
      pw[j + hh * 8][16 + m] = f2bf(p1);
    }
    __syncthreads();
    v16bf ap = pack16(*(const bf16x8*)(&pw[m][kb]),
                      *(const bf16x8*)(&pw[m][kb + 8]));
    __syncthreads();

    int c0 = kstart + hh * 16;
    if (c0 < 0) c0 = 0;
    if (c0 > CS - 16) c0 = CS - 16;
#pragma unroll
    for (int t = 0; t < 4; ++t) {
      const __bf16* vrow = Vp + (size_t)(t * 16 + m) * CS + c0;
      v16bf bv = pack16(*(const bf16x8*)(vrow), *(const bf16x8*)(vrow + 8));
      oacc[t] = wmma_bf16(ap, bv, oacc[t]);
    }
  }

#pragma unroll
  for (int j = 0; j < 8; ++j) {
    float t = tacc[j];
    t += __shfl_xor(t, 1); t += __shfl_xor(t, 2);
    t += __shfl_xor(t, 4); t += __shfl_xor(t, 8);
    float inv = 1.0f / t;
    int srow = q0 + j + hh * 8;
    __bf16* orow = Ob + ((size_t)bb * CS + srow) * CD + h * CDh;
#pragma unroll
    for (int t4 = 0; t4 < 4; ++t4)
      orow[t4 * 16 + m] = f2bf(oacc[t4][j] * inv);
  }
}

// ---------------------------------------------------------------------------
// Kernel 4: output projection, staged via async global->LDS copies (ASYNCcnt)
// ---------------------------------------------------------------------------
__global__ __launch_bounds__(128) void out_proj(
    const __bf16* __restrict__ Xb, const __bf16* __restrict__ Wo,
    const float* __restrict__ bo, float* __restrict__ out) {
  __shared__ __align__(16) __bf16 xs[16 * CD];
  const int tid = threadIdx.x;
  const int rowtile = blockIdx.x;
  const __bf16* xsrc = Xb + (size_t)rowtile * 16 * CD;

  // async DMA: 128 lanes x 16B x 16 iters = 32 KB, no VGPR round-trip
  {
    uint64_t sbase = (uint64_t)(uintptr_t)xsrc;
    uint32_t lbase = (uint32_t)(uintptr_t)xs;
    int off = tid * 16;
#pragma unroll
    for (int i = 0; i < 16; ++i) {
      uint32_t la = lbase + off + i * 2048;
      int      go = off + i * 2048;
      asm volatile("global_load_async_to_lds_b128 %0, %1, %2"
                   :: "v"(la), "v"(go), "s"(sbase) : "memory");
    }
    asm volatile("s_wait_asynccnt 0x0" ::: "memory");
  }
  __syncthreads();

  const int lane = tid & 31, wave = tid >> 5;
  const int colbase = (blockIdx.y * 4 + wave) * 64;
  const int m = lane & 15, hh = lane >> 4, kb = hh * 16;
  const __bf16* wbase = Wo + (size_t)(colbase + m) * CD + hh * 16;

  v8f acc[4] = {};
  for (int k0 = 0; k0 < CD; k0 += 32) {
    bf16x8 alo = *(const bf16x8*)(xs + m * CD + k0 + kb);
    bf16x8 ahi = *(const bf16x8*)(xs + m * CD + k0 + kb + 8);
    v16bf a = pack16(alo, ahi);
    __builtin_prefetch(wbase + k0 + 128, 0, 3);
#pragma unroll
    for (int t = 0; t < 4; ++t) {
      const __bf16* wrow = wbase + (size_t)t * 16 * CD + k0;
      v16bf b = pack16(*(const bf16x8*)(wrow), *(const bf16x8*)(wrow + 8));
      acc[t] = wmma_bf16(a, b, acc[t]);
    }
  }

  const int rowbase = rowtile * 16;
#pragma unroll
  for (int t = 0; t < 4; ++t) {
    const int o = colbase + t * 16 + m;
    const float bias = bo[o];
#pragma unroll
    for (int j = 0; j < 8; ++j)
      out[(size_t)(rowbase + j + hh * 8) * CD + o] = acc[t][j] + bias;
  }
}

// ---------------------------------------------------------------------------
extern "C" void kernel_launch(void* const* d_in, const int* in_sizes, int n_in,
                              void* d_out, int out_size, void* d_ws,
                              size_t ws_size, hipStream_t stream) {
  const float* x  = (const float*)d_in[0];
  const float* wq = (const float*)d_in[1];
  const float* bq = (const float*)d_in[2];
  const float* wk = (const float*)d_in[3];
  const float* bk = (const float*)d_in[4];
  const float* wv = (const float*)d_in[5];
  const float* bv = (const float*)d_in[6];
  const float* wo = (const float*)d_in[7];
  const float* bo = (const float*)d_in[8];
  float* out = (float*)d_out;

  char* ws = (char*)d_ws;
  size_t off = 0;
  auto alloc = [&](size_t bytes) -> char* {
    char* p = ws + off;
    off += (bytes + 255) & ~(size_t)255;
    return p;
  };
  __bf16* Xbf = (__bf16*)alloc((size_t)CM * CD * 2);
  __bf16* Wqb = (__bf16*)alloc((size_t)CD * CD * 2);
  __bf16* Wkb = (__bf16*)alloc((size_t)CD * CD * 2);
  __bf16* Wvb = (__bf16*)alloc((size_t)CD * CD * 2);
  __bf16* Wob = (__bf16*)alloc((size_t)CD * CD * 2);
  __bf16* Qs  = (__bf16*)alloc((size_t)CM * CD * 2);
  __bf16* Kbf = (__bf16*)alloc((size_t)CM * CD * 2);
  __bf16* Vt  = (__bf16*)alloc((size_t)CM * CD * 2);
  __bf16* Ob  = (__bf16*)alloc((size_t)CM * CD * 2);

  cvt_f32_bf16<<<(CM * CD) / 2048, 256, 0, stream>>>(x, Xbf, CM * CD);
  cvt_f32_bf16<<<(CD * CD) / 2048, 256, 0, stream>>>(wq, Wqb, CD * CD);
  cvt_f32_bf16<<<(CD * CD) / 2048, 256, 0, stream>>>(wk, Wkb, CD * CD);
  cvt_f32_bf16<<<(CD * CD) / 2048, 256, 0, stream>>>(wv, Wvb, CD * CD);
  cvt_f32_bf16<<<(CD * CD) / 2048, 256, 0, stream>>>(wo, Wob, CD * CD);

  qkv_proj<<<dim3(CM / 16, CD / 256), 128, 0, stream>>>(
      Xbf, Wqb, Wkb, Wvb, bq, bk, bv, Qs, Kbf, Vt);

  attn_win<<<dim3(CS / 64, CB * CH), 128, 0, stream>>>(Qs, Kbf, Vt, Ob);

  out_proj<<<dim3(CM / 16, CD / 256), 128, 0, stream>>>(Ob, Wob, bo, out);
}
